// MambaBlock_29300266893413
// MI455X (gfx1250) — compile-verified, matrix operand order checked by reading, not run on hardware
//
#include <hip/hip_runtime.h>

// ---------------- problem constants ----------------
constexpr int D_MODEL = 1024;
constexpr int D_INNER = 2048;
constexpr int D_STATE = 16;
constexpr int DT_RANK = 64;
constexpr int D_CONV  = 4;
constexpr int BB      = 2;
constexpr int LSEQ    = 2048;
constexpr int T       = BB * LSEQ;             // 4096 tokens
constexpr int XPROJ_N = DT_RANK + 2 * D_STATE; // 96

// ---------------- WMMA fragment types --------------
typedef __attribute__((ext_vector_type(16))) __bf16 v16bf;
typedef __attribute__((ext_vector_type(8)))  float  v8f;

__device__ __forceinline__ unsigned short f2bf(float f) {
  unsigned int u = __float_as_uint(f);
  u += 0x7FFFu + ((u >> 16) & 1u);      // round-to-nearest-even
  return (unsigned short)(u >> 16);
}

__device__ __forceinline__ float silu(float x) {
  return x / (1.0f + __expf(-x));
}

union FragCast { uint4 q[2]; v16bf v; };

// Load one 16x32 bf16 WMMA operand fragment from a row-major panel.
// ISA 16-bit A layout: lanes 0-15 -> M=lane, K = {0..7, 16..23};
//                      lanes 16-31 -> M=lane-16, K = {8..15, 24..31}.
__device__ __forceinline__ v16bf load_frag(const unsigned short* __restrict__ P,
                                           int ld, int row0, int k0, int lane) {
  const int r  = row0 + (lane & 15);
  const int kb = k0 + ((lane >> 4) << 3);
  const unsigned short* p = P + (size_t)r * ld + kb;
  FragCast f;
  f.q[0] = *reinterpret_cast<const uint4*>(p);
  f.q[1] = *reinterpret_cast<const uint4*>(p + 16);
  return f.v;
}

// LDS variant: rows padded to 40 elements (80B stride -> conflict-free banks,
// 16B-aligned fragment chunks).
constexpr int LDSP = 40;
__device__ __forceinline__ v16bf frag_lds(const unsigned short (*S)[LDSP],
                                          int row0, int lane) {
  const int r  = row0 + (lane & 15);
  const int kb = (lane >> 4) << 3;
  FragCast f;
  f.q[0] = *reinterpret_cast<const uint4*>(&S[r][kb]);
  f.q[1] = *reinterpret_cast<const uint4*>(&S[r][kb + 16]);
  return f.v;
}

// CDNA5 async global->LDS copy of one 16-byte chunk (tracked by ASYNCcnt).
__device__ __forceinline__ void async_g2l_b128(const unsigned short* gptr,
                                               const unsigned short* lptr) {
  unsigned lds_addr = (unsigned)(reinterpret_cast<uintptr_t>(lptr));
  unsigned long long gaddr = (unsigned long long)(reinterpret_cast<uintptr_t>(gptr));
  asm volatile("global_load_async_to_lds_b128 %0, %1, off"
               :: "v"(lds_addr), "v"(gaddr) : "memory");
}
__device__ __forceinline__ void wait_async0() {
  asm volatile("s_wait_asynccnt 0x0" ::: "memory");
}

// ---------------- elementwise helpers --------------
__global__ void k_cvt_bf16(const float* __restrict__ in,
                           unsigned short* __restrict__ out, int n) {
  int i = blockIdx.x * blockDim.x + threadIdx.x;
  if (i < n) out[i] = f2bf(in[i]);
}

// in[R x C] row-major f32 -> out[C x R] row-major bf16
__global__ void k_transpose_bf16(const float* __restrict__ in,
                                 unsigned short* __restrict__ out, int R, int C) {
  int i = blockIdx.x * blockDim.x + threadIdx.x;
  if (i >= R * C) return;
  int r = i / C, c = i - r * C;
  out[(size_t)c * R + r] = f2bf(in[i]);
}

// ---------------- direct bf16 WMMA GEMM (small-N path) ------------
// C[M x N] (f32) = A[M x K] bf16 row-major * Bt[N x K]^T bf16 row-major.
// One wave computes a 32x32 tile. M,N,K multiples of 32.
__global__ void k_gemm_bf16(const unsigned short* __restrict__ A,
                            const unsigned short* __restrict__ Bt,
                            float* __restrict__ C, int M, int N, int K) {
  const int lane   = threadIdx.x & 31;
  const int wave   = (blockIdx.x * blockDim.x + threadIdx.x) >> 5;
  const int tilesN = N >> 5;
  const int tilesM = M >> 5;
  const int tm = wave / tilesN;
  const int tn = wave - tm * tilesN;
  if (tm >= tilesM) return;                 // wave-uniform: EXEC all-1 for WMMA
  const int m0 = tm << 5, n0 = tn << 5;

  v8f c00 = {}, c01 = {}, c10 = {}, c11 = {};
  for (int k = 0; k < K; k += 32) {
    v16bf a0 = load_frag(A,  K, m0,      k, lane);
    v16bf a1 = load_frag(A,  K, m0 + 16, k, lane);
    v16bf b0 = load_frag(Bt, K, n0,      k, lane);
    v16bf b1 = load_frag(Bt, K, n0 + 16, k, lane);
    if (k + 32 < K) {
      __builtin_prefetch(A  + (size_t)(m0 + (lane & 15)) * K + k + 32, 0, 1);
      __builtin_prefetch(Bt + (size_t)(n0 + (lane & 15)) * K + k + 32, 0, 1);
    }
    c00 = __builtin_amdgcn_wmma_f32_16x16x32_bf16(false, a0, false, b0, (short)0, c00, false, false);
    c01 = __builtin_amdgcn_wmma_f32_16x16x32_bf16(false, a0, false, b1, (short)0, c01, false, false);
    c10 = __builtin_amdgcn_wmma_f32_16x16x32_bf16(false, a1, false, b0, (short)0, c10, false, false);
    c11 = __builtin_amdgcn_wmma_f32_16x16x32_bf16(false, a1, false, b1, (short)0, c11, false, false);
  }

  const int col  = lane & 15;
  const int radd = (lane >> 4) << 3;
  float* Cp = C + (size_t)(m0 + radd) * N + n0 + col;
  #pragma unroll
  for (int v = 0; v < 8; ++v) {
    Cp[(size_t)v * N]             = c00[v];
    Cp[(size_t)v * N + 16]        = c01[v];
    Cp[(size_t)(v + 16) * N]      = c10[v];
    Cp[(size_t)(v + 16) * N + 16] = c11[v];
  }
}

// ---------------- LDS-staged bf16 WMMA GEMM (big GEMMs) -----------
// Block = 256 threads (8 waves), block tile 128(M) x 64(N), k-slice 32.
// A/B k-slices staged in LDS via async global->LDS b128 copies, double
// buffered; each wave computes a 32x32 tile (waves arranged 4x2).
// Requires: M % 128 == 0, N % 64 == 0, K % 32 == 0.
__global__ void __launch_bounds__(256)
k_gemm_bf16_lds(const unsigned short* __restrict__ A,
                const unsigned short* __restrict__ Bt,
                float* __restrict__ C, int M, int N, int K) {
  __shared__ __align__(16) unsigned short As[2][128][LDSP];
  __shared__ __align__(16) unsigned short Bs[2][64][LDSP];

  const int tilesN = N >> 6;
  const int tm = blockIdx.x / tilesN;
  const int tn = blockIdx.x - tm * tilesN;
  const int m0 = tm << 7;                 // block row base
  const int n0 = tn << 6;                 // block col base
  const int tid  = threadIdx.x;
  const int lane = tid & 31;
  const int wv   = tid >> 5;              // 0..7
  const int wm   = (wv >> 1) << 5;        // wave row offset in block tile
  const int wn   = (wv & 1) << 5;         // wave col offset in block tile

  // Stage one 32-wide k-slice of A (128x32, 512 chunks) and Bt (64x32, 256).
  auto stageA = [&](int buf, int k0) {
    for (int c = tid; c < 512; c += 256) {
      int row = c >> 2, kc = (c & 3) << 3;
      async_g2l_b128(A + (size_t)(m0 + row) * K + k0 + kc, &As[buf][row][kc]);
    }
  };
  auto stageB = [&](int buf, int k0) {
    int row = tid >> 2, kc = (tid & 3) << 3;
    async_g2l_b128(Bt + (size_t)(n0 + row) * K + k0 + kc, &Bs[buf][row][kc]);
  };

  stageA(0, 0);
  stageB(0, 0);
  wait_async0();
  __syncthreads();

  v8f c00 = {}, c01 = {}, c10 = {}, c11 = {};
  for (int k = 0; k < K; k += 32) {
    const int cur = (k >> 5) & 1;
    const int nxt = cur ^ 1;
    if (k + 32 < K) {                     // prefetch next slice into other buf
      stageA(nxt, k + 32);
      stageB(nxt, k + 32);
    }
    v16bf a0 = frag_lds(As[cur], wm,      lane);
    v16bf a1 = frag_lds(As[cur], wm + 16, lane);
    v16bf b0 = frag_lds(Bs[cur], wn,      lane);
    v16bf b1 = frag_lds(Bs[cur], wn + 16, lane);
    c00 = __builtin_amdgcn_wmma_f32_16x16x32_bf16(false, a0, false, b0, (short)0, c00, false, false);
    c01 = __builtin_amdgcn_wmma_f32_16x16x32_bf16(false, a0, false, b1, (short)0, c01, false, false);
    c10 = __builtin_amdgcn_wmma_f32_16x16x32_bf16(false, a1, false, b0, (short)0, c10, false, false);
    c11 = __builtin_amdgcn_wmma_f32_16x16x32_bf16(false, a1, false, b1, (short)0, c11, false, false);
    wait_async0();                        // next slice resident in LDS
    __syncthreads();                      // all waves done with cur & see nxt
  }

  const int col  = lane & 15;
  const int radd = (lane >> 4) << 3;
  float* Cp = C + (size_t)(m0 + wm + radd) * N + n0 + wn + col;
  #pragma unroll
  for (int v = 0; v < 8; ++v) {
    Cp[(size_t)v * N]             = c00[v];
    Cp[(size_t)v * N + 16]        = c01[v];
    Cp[(size_t)(v + 16) * N]      = c10[v];
    Cp[(size_t)(v + 16) * N + 16] = c11[v];
  }
}

// ---------------- depthwise causal conv + SiLU -----
__global__ void k_conv_silu(const float* __restrict__ xz,
                            const float* __restrict__ conv_w,
                            const float* __restrict__ conv_b,
                            float* __restrict__ u,
                            unsigned short* __restrict__ u_bf) {
  int i = blockIdx.x * blockDim.x + threadIdx.x;   // over T * D_INNER (exact)
  int t = i / D_INNER, d = i - t * D_INNER;
  int lt = t & (LSEQ - 1);
  float acc = conv_b[d];
  #pragma unroll
  for (int j = 0; j < D_CONV; ++j) {
    int dt = lt - (D_CONV - 1) + j;
    if (dt >= 0)
      acc += xz[(size_t)(t - (D_CONV - 1) + j) * (2 * D_INNER) + d] * conv_w[d * D_CONV + j];
  }
  float uu = silu(acc);
  u[i]    = uu;
  u_bf[i] = f2bf(uu);
}

// dt_bf[t, 0:64] = bf16(xdbl[t, 0:64])
__global__ void k_dt_extract(const float* __restrict__ xdbl,
                             unsigned short* __restrict__ dt_bf) {
  int i = blockIdx.x * blockDim.x + threadIdx.x;   // over T * DT_RANK (exact)
  int t = i / DT_RANK, j = i - t * DT_RANK;
  dt_bf[i] = f2bf(xdbl[(size_t)t * XPROJ_N + j]);
}

// delta = softplus(delta_raw + b_dt), in place
__global__ void k_softplus_bias(float* __restrict__ dlt,
                                const float* __restrict__ b_dt) {
  int i = blockIdx.x * blockDim.x + threadIdx.x;   // over T * D_INNER (exact)
  int d = i & (D_INNER - 1);
  float v = dlt[i] + b_dt[d];
  dlt[i] = (v > 20.0f) ? v : log1pf(__expf(v));
}

// ---------------- fused selective scan + gate ------
__global__ void k_scan_gate(const float* __restrict__ u,
                            const float* __restrict__ dlt,
                            const float* __restrict__ xdbl,
                            const float* __restrict__ A_log,
                            const float* __restrict__ Dvec,
                            const float* __restrict__ xz,
                            unsigned short* __restrict__ y_bf) {
  int gid  = blockIdx.x * blockDim.x + threadIdx.x;   // BB*D_INNER*16 lanes exact
  int s    = gid & (D_STATE - 1);
  int grp  = gid >> 4;                                // (b, ch)
  int b    = grp / D_INNER;
  int ch   = grp - b * D_INNER;

  const float Ac = -__expf(A_log[ch * D_STATE + s]);
  const float Dd = Dvec[ch];
  float h = 0.0f;

  for (int lt = 0; lt < LSEQ; ++lt) {
    int t = b * LSEQ + lt;
    float dl = dlt[(size_t)t * D_INNER + ch];
    float ut = u  [(size_t)t * D_INNER + ch];
    float Bs = xdbl[(size_t)t * XPROJ_N + DT_RANK + s];
    float Cs = xdbl[(size_t)t * XPROJ_N + DT_RANK + D_STATE + s];
    float dA = __expf(dl * Ac);
    h = h * dA + dl * Bs * ut;
    float p = h * Cs;
    p += __shfl_xor(p, 8, 16);
    p += __shfl_xor(p, 4, 16);
    p += __shfl_xor(p, 2, 16);
    p += __shfl_xor(p, 1, 16);
    if (s == 0) {
      float zv = xz[(size_t)t * (2 * D_INNER) + D_INNER + ch];
      float yv = (p + ut * Dd) * silu(zv);
      y_bf[(size_t)t * D_INNER + ch] = f2bf(yv);
    }
  }
}

// ---------------- host launcher --------------------
extern "C" void kernel_launch(void* const* d_in, const int* in_sizes, int n_in,
                              void* d_out, int out_size, void* d_ws, size_t ws_size,
                              hipStream_t stream) {
  const float* x      = (const float*)d_in[0];
  const float* W_in   = (const float*)d_in[1];
  const float* conv_w = (const float*)d_in[2];
  const float* conv_b = (const float*)d_in[3];
  const float* W_xprj = (const float*)d_in[4];
  const float* W_dt   = (const float*)d_in[5];
  const float* b_dt   = (const float*)d_in[6];
  const float* A_log  = (const float*)d_in[7];
  const float* Dvec   = (const float*)d_in[8];
  const float* W_out  = (const float*)d_in[9];
  float* out = (float*)d_out;

  // workspace arena (256B-aligned slabs)
  char* w = (char*)d_ws;
  auto alloc = [&](size_t bytes) -> void* {
    void* p = (void*)w;
    w += (bytes + 255) & ~(size_t)255;
    return p;
  };
  unsigned short* x_bf  = (unsigned short*)alloc((size_t)T * D_MODEL * 2);
  unsigned short* Wi_t  = (unsigned short*)alloc((size_t)(2 * D_INNER) * D_MODEL * 2);
  float*          xz    = (float*)        alloc((size_t)T * 2 * D_INNER * 4);
  float*          u     = (float*)        alloc((size_t)T * D_INNER * 4);
  unsigned short* u_bf  = (unsigned short*)alloc((size_t)T * D_INNER * 2);
  unsigned short* Wx_t  = (unsigned short*)alloc((size_t)XPROJ_N * D_INNER * 2);
  float*          xdbl  = (float*)        alloc((size_t)T * XPROJ_N * 4);
  unsigned short* dt_bf = (unsigned short*)alloc((size_t)T * DT_RANK * 2);
  unsigned short* Wdt_t = (unsigned short*)alloc((size_t)D_INNER * DT_RANK * 2);
  float*          dlt   = (float*)        alloc((size_t)T * D_INNER * 4);
  unsigned short* y_bf  = (unsigned short*)alloc((size_t)T * D_INNER * 2);
  unsigned short* Wo_t  = (unsigned short*)alloc((size_t)D_MODEL * D_INNER * 2);

  const int EB = 256;
  auto eg = [](size_t n) { return (unsigned)((n + 255) / 256); };

  // --- stage 0: precision conversion / weight transposition ---
  k_cvt_bf16<<<eg((size_t)T * D_MODEL), EB, 0, stream>>>(x, x_bf, T * D_MODEL);
  k_transpose_bf16<<<eg((size_t)D_MODEL * 2 * D_INNER), EB, 0, stream>>>(W_in,  Wi_t,  D_MODEL, 2 * D_INNER);
  k_transpose_bf16<<<eg((size_t)D_INNER * XPROJ_N),     EB, 0, stream>>>(W_xprj, Wx_t, D_INNER, XPROJ_N);
  k_transpose_bf16<<<eg((size_t)DT_RANK * D_INNER),     EB, 0, stream>>>(W_dt,  Wdt_t, DT_RANK, D_INNER);
  k_transpose_bf16<<<eg((size_t)D_INNER * D_MODEL),     EB, 0, stream>>>(W_out, Wo_t,  D_INNER, D_MODEL);

  auto lds_grid = [](int M, int N) { return (unsigned)((M >> 7) * (N >> 6)); };

  // --- stage 1: xz = x @ W_in  (4096x4096x1024, LDS-staged async) ---
  k_gemm_bf16_lds<<<lds_grid(T, 2 * D_INNER), 256, 0, stream>>>(x_bf, Wi_t, xz, T, 2 * D_INNER, D_MODEL);

  // --- stage 2: depthwise causal conv + SiLU -> u ---
  k_conv_silu<<<eg((size_t)T * D_INNER), EB, 0, stream>>>(xz, conv_w, conv_b, u, u_bf);

  // --- stage 3: x_dbl = u @ W_xproj  (N=96, direct wave-tile path) ---
  {
    unsigned waves = (unsigned)((T >> 5) * (XPROJ_N >> 5));
    k_gemm_bf16<<<(waves + 3) / 4, 128, 0, stream>>>(u_bf, Wx_t, xdbl, T, XPROJ_N, D_INNER);
  }

  // --- stage 4: delta = softplus(dt @ W_dt + b_dt)  (K=64) ---
  k_dt_extract<<<eg((size_t)T * DT_RANK), EB, 0, stream>>>(xdbl, dt_bf);
  k_gemm_bf16_lds<<<lds_grid(T, D_INNER), 256, 0, stream>>>(dt_bf, Wdt_t, dlt, T, D_INNER, DT_RANK);
  k_softplus_bias<<<eg((size_t)T * D_INNER), EB, 0, stream>>>(dlt, b_dt);

  // --- stage 5: fused selective scan + u*D + SiLU gate -> y (bf16) ---
  k_scan_gate<<<(BB * D_INNER * D_STATE) / EB, EB, 0, stream>>>(u, dlt, xdbl, A_log, Dvec, xz, y_bf);

  // --- stage 6: out = y @ W_out  (f32 straight to d_out) ---
  k_gemm_bf16_lds<<<lds_grid(T, D_MODEL), 256, 0, stream>>>(y_bf, Wo_t, out, T, D_MODEL, D_INNER);
}
